// CrossAttention_6717328851348
// MI455X (gfx1250) — compile-verified, matrix operand order checked by reading, not run on hardware
//
#include <hip/hip_runtime.h>
#include <cstdint>
#include <cstddef>

// ---------------- problem constants ----------------
#define BB 4
#define NQ 4096
#define NK 1024
#define QD 1024
#define CD 768
#define NH 16
#define HD 64
#define ATT_SCALE 0.125f  // 64^-0.5

// ---------------- WMMA types ----------------
typedef __attribute__((ext_vector_type(16))) __bf16 v16bf;
typedef __attribute__((ext_vector_type(8)))  __bf16 v8bf;
typedef __attribute__((ext_vector_type(8)))  float  v8f;

static __device__ inline __bf16 f2bf(float f) {
  union { float f; unsigned u; } in;
  in.f = f;
  unsigned r = in.u + 0x7FFFu + ((in.u >> 16) & 1u);  // RNE
  union { unsigned short s; __bf16 b; } out;
  out.s = (unsigned short)(r >> 16);
  return out.b;
}

union FragU { v16bf v; v8bf h[2]; };

// A-fragment (16x32, MxK) and B-fragment (from Bt stored [N][K]) share the
// same per-lane pattern: lane l -> row/col (l&15), k = (l>>4)*8 + {0..7,16..23}
static __device__ inline v16bf load_frag(const __bf16* base, int stride, int lane) {
  const __bf16* p = base + (size_t)(lane & 15) * stride + ((lane >> 4) << 3);
  FragU f;
  f.h[0] = *reinterpret_cast<const v8bf*>(p);
  f.h[1] = *reinterpret_cast<const v8bf*>(p + 16);
  return f.v;
}

static __device__ inline v8f wmma_bf(v16bf a, v16bf b, v8f c) {
  return __builtin_amdgcn_wmma_f32_16x16x32_bf16(false, a, false, b, (short)0, c,
                                                 false, false);
}

#define V8F_ZERO {0.f,0.f,0.f,0.f,0.f,0.f,0.f,0.f}

// ---------------- conversion kernels ----------------
__global__ void cvt_f32_bf16(const float* __restrict__ in, __bf16* __restrict__ out,
                             int n) {
  int i = blockIdx.x * blockDim.x + threadIdx.x;
  if (i < n) out[i] = f2bf(in[i]);
}

// Wt[n*K + k] = W[k*N + n]  (f32 -> bf16)
__global__ void transpose_cvt(const float* __restrict__ W, __bf16* __restrict__ Wt,
                              int K, int N) {
  int i = blockIdx.x * blockDim.x + threadIdx.x;
  if (i < K * N) {
    int n = i / K;
    int k = i % K;
    Wt[i] = f2bf(W[(size_t)k * N + n]);
  }
}

// ---------------- bf16 WMMA GEMM: C = A @ Bt^T, 16x64 tile per wave ----------
// A: [M][lda] bf16 row-major, Bt: [N][ldb] bf16 (i.e. B transposed).
// N64tiles = N/64. Each wave: rows tm*16..+15, cols tn64*64..+63.
// mode 0: out bf16 [M][ldo]. mode 1: out = Vt layout [b][h][d][key] (M=B*NK, N=QD).
__global__ void __launch_bounds__(256)
gemm_bf16(const __bf16* __restrict__ A, const __bf16* __restrict__ Bt,
          __bf16* __restrict__ out, int N64tiles, int Kdim, int lda, int ldb,
          int ldo, int mode) {
  int lane = threadIdx.x & 31;
  int wave = threadIdx.x >> 5;
  int tile = blockIdx.x * 8 + wave;
  int tm = tile / N64tiles;
  int tn64 = tile % N64tiles;
  const __bf16* arow = A + (size_t)tm * 16 * lda;
  const __bf16* b0r = Bt + (size_t)(tn64 * 64) * ldb;
  const __bf16* b1r = b0r + (size_t)16 * ldb;
  const __bf16* b2r = b0r + (size_t)32 * ldb;
  const __bf16* b3r = b0r + (size_t)48 * ldb;
  v8f c0 = V8F_ZERO, c1 = V8F_ZERO, c2 = V8F_ZERO, c3 = V8F_ZERO;
  for (int kk = 0; kk < Kdim; kk += 32) {
    v16bf a = load_frag(arow + kk, lda, lane);  // reused across 4 N-tiles
    c0 = wmma_bf(a, load_frag(b0r + kk, ldb, lane), c0);
    c1 = wmma_bf(a, load_frag(b1r + kk, ldb, lane), c1);
    c2 = wmma_bf(a, load_frag(b2r + kk, ldb, lane), c2);
    c3 = wmma_bf(a, load_frag(b3r + kk, ldb, lane), c3);
  }
  int nn = lane & 15;
  int hl = lane >> 4;
#pragma unroll
  for (int t = 0; t < 4; ++t) {
    v8f c = (t == 0) ? c0 : (t == 1) ? c1 : (t == 2) ? c2 : c3;
#pragma unroll
    for (int r = 0; r < 8; ++r) {
      int m = r + hl * 8;
      int gm = tm * 16 + m;
      int gn = tn64 * 64 + t * 16 + nn;
      if (mode == 0) {
        out[(size_t)gm * ldo + gn] = f2bf(c[r]);
      } else {  // V transposed store: [b][h][d][key], key dim = NK = 1024
        int b_ = gm >> 10, key = gm & 1023;
        int h = gn >> 6, d = gn & 63;
        out[(((size_t)(b_ * NH + h) * HD + d) << 10) + key] = f2bf(c[r]);
      }
    }
  }
}

// ---------------- final GEMM: out(f32) = Ab @ WoT^T + bias, 16x64/wave ------
__global__ void __launch_bounds__(256)
gemm_out(const __bf16* __restrict__ A, const __bf16* __restrict__ Bt,
         const float* __restrict__ bias, float* __restrict__ out, int N64tiles,
         int Kdim) {
  int lane = threadIdx.x & 31;
  int wave = threadIdx.x >> 5;
  int tile = blockIdx.x * 8 + wave;
  int tm = tile / N64tiles;
  int tn64 = tile % N64tiles;
  const __bf16* arow = A + (size_t)tm * 16 * Kdim;
  const __bf16* b0r = Bt + (size_t)(tn64 * 64) * Kdim;
  const __bf16* b1r = b0r + (size_t)16 * Kdim;
  const __bf16* b2r = b0r + (size_t)32 * Kdim;
  const __bf16* b3r = b0r + (size_t)48 * Kdim;
  v8f c0 = V8F_ZERO, c1 = V8F_ZERO, c2 = V8F_ZERO, c3 = V8F_ZERO;
  for (int kk = 0; kk < Kdim; kk += 32) {
    v16bf a = load_frag(arow + kk, Kdim, lane);
    c0 = wmma_bf(a, load_frag(b0r + kk, Kdim, lane), c0);
    c1 = wmma_bf(a, load_frag(b1r + kk, Kdim, lane), c1);
    c2 = wmma_bf(a, load_frag(b2r + kk, Kdim, lane), c2);
    c3 = wmma_bf(a, load_frag(b3r + kk, Kdim, lane), c3);
  }
  int nn = lane & 15;
  int hl = lane >> 4;
  int Ncols = N64tiles * 64;
#pragma unroll
  for (int t = 0; t < 4; ++t) {
    v8f c = (t == 0) ? c0 : (t == 1) ? c1 : (t == 2) ? c2 : c3;
#pragma unroll
    for (int r = 0; r < 8; ++r) {
      int m = r + hl * 8;
      int gm = tm * 16 + m;
      int gn = tn64 * 64 + t * 16 + nn;
      out[(size_t)gm * Ncols + gn] = c[r] + bias[gn];
    }
  }
}

// ---------------- flash attention ----------------
// grid (NQ/16/8, NH, BB), block 256 (8 waves). Each wave: one 16-row q tile.
// Qb: [B*NQ][QD] bf16 ; Kb: [B*NK][QD] bf16 ; Vt: [B][H][HD][NK] bf16
// Ab: [B*NQ][QD] bf16 (heads interleaved, matching transpose(0,2,1,3).reshape)
__global__ void __launch_bounds__(256)
flash_attn(const __bf16* __restrict__ Qb, const __bf16* __restrict__ Kb,
           const __bf16* __restrict__ Vt, __bf16* __restrict__ Ab) {
  __shared__ __align__(16) __bf16 smem[8 * 16 * 32];
  int lane = threadIdx.x & 31;
  int wave = threadIdx.x >> 5;
  int b = blockIdx.z;
  int h = blockIdx.y;
  int qbase = (blockIdx.x * 8 + wave) * 16;

  const __bf16* qrow = Qb + (size_t)(b * NQ + qbase) * QD + h * HD;
  v16bf aq0 = load_frag(qrow, QD, lane);       // k = 0..31 of HD
  v16bf aq1 = load_frag(qrow + 32, QD, lane);  // k = 32..63

  float mr[8], lr[8];
#pragma unroll
  for (int r = 0; r < 8; ++r) { mr[r] = -__builtin_inff(); lr[r] = 0.f; }
  v8f o0 = V8F_ZERO, o1 = V8F_ZERO, o2 = V8F_ZERO, o3 = V8F_ZERO;

  int hl = lane >> 4;
  int nn = lane & 15;
  __bf16* sm = smem + wave * (16 * 32);
  const __bf16* vbase = Vt + (size_t)(b * NH + h) * HD * NK;

  for (int kt = 0; kt < NK; kt += 32) {
    // scores: S = Q (16xHD) @ K^T, two 16-key tiles
    const __bf16* kr = Kb + (size_t)(b * NK + kt) * QD + h * HD;
    v8f s0 = V8F_ZERO, s1 = V8F_ZERO;
    s0 = wmma_bf(aq0, load_frag(kr, QD, lane), s0);
    s0 = wmma_bf(aq1, load_frag(kr + 32, QD, lane), s0);
    kr += (size_t)16 * QD;
    s1 = wmma_bf(aq0, load_frag(kr, QD, lane), s1);
    s1 = wmma_bf(aq1, load_frag(kr + 32, QD, lane), s1);

    float alpha[8], p0[8], p1[8];
#pragma unroll
    for (int r = 0; r < 8; ++r) {
      float a0 = s0[r] * ATT_SCALE;
      float a1 = s1[r] * ATT_SCALE;
      // row max across the 16 lanes of this half (row = r + 8*hl)
      float mx = fmaxf(a0, a1);
      mx = fmaxf(mx, __shfl_xor(mx, 1, 32));
      mx = fmaxf(mx, __shfl_xor(mx, 2, 32));
      mx = fmaxf(mx, __shfl_xor(mx, 4, 32));
      mx = fmaxf(mx, __shfl_xor(mx, 8, 32));
      float mnew = fmaxf(mr[r], mx);
      alpha[r] = __expf(mr[r] - mnew);
      p0[r] = __expf(a0 - mnew);
      p1[r] = __expf(a1 - mnew);
      float sum = p0[r] + p1[r];
      sum += __shfl_xor(sum, 1, 32);
      sum += __shfl_xor(sum, 2, 32);
      sum += __shfl_xor(sum, 4, 32);
      sum += __shfl_xor(sum, 8, 32);
      lr[r] = lr[r] * alpha[r] + sum;
      mr[r] = mnew;
    }
    // rescale O accumulators, stage P tile (C-layout -> LDS row-major 16x32)
#pragma unroll
    for (int r = 0; r < 8; ++r) {
      o0[r] *= alpha[r];
      o1[r] *= alpha[r];
      o2[r] *= alpha[r];
      o3[r] *= alpha[r];
      int m = r + hl * 8;
      sm[m * 32 + nn] = f2bf(p0[r]);
      sm[m * 32 + 16 + nn] = f2bf(p1[r]);
    }
    __syncthreads();
    v16bf pf = load_frag(sm, 32, lane);  // P as A-fragment 16x32
    __syncthreads();
    // O += P @ V_tile : V_tile is 32(keys) x 64(hd) -> 4 column tiles
    o0 = wmma_bf(pf, load_frag(vbase + kt, NK, lane), o0);
    o1 = wmma_bf(pf, load_frag(vbase + (size_t)16 * NK + kt, NK, lane), o1);
    o2 = wmma_bf(pf, load_frag(vbase + (size_t)32 * NK + kt, NK, lane), o2);
    o3 = wmma_bf(pf, load_frag(vbase + (size_t)48 * NK + kt, NK, lane), o3);
  }

  __bf16* orow = Ab + (size_t)(b * NQ + qbase) * QD + h * HD;
#pragma unroll
  for (int r = 0; r < 8; ++r) {
    int m = r + hl * 8;
    float inv = 1.f / lr[r];
    orow[(size_t)m * QD + nn]      = f2bf(o0[r] * inv);
    orow[(size_t)m * QD + 16 + nn] = f2bf(o1[r] * inv);
    orow[(size_t)m * QD + 32 + nn] = f2bf(o2[r] * inv);
    orow[(size_t)m * QD + 48 + nn] = f2bf(o3[r] * inv);
  }
}

// ---------------- launcher ----------------
extern "C" void kernel_launch(void* const* d_in, const int* in_sizes, int n_in,
                              void* d_out, int out_size, void* d_ws, size_t ws_size,
                              hipStream_t stream) {
  const float* x   = (const float*)d_in[0];
  const float* ctx = (const float*)d_in[1];
  const float* Wq  = (const float*)d_in[2];
  const float* Wk  = (const float*)d_in[3];
  const float* Wv  = (const float*)d_in[4];
  const float* Wo  = (const float*)d_in[5];
  const float* bo  = (const float*)d_in[6];
  float* out = (float*)d_out;

  // workspace carve (bf16 elements)
  __bf16* p = (__bf16*)d_ws;
  __bf16* xb  = p; p += (size_t)BB * NQ * QD;   // x bf16
  __bf16* cb  = p; p += (size_t)BB * NK * CD;   // context bf16
  __bf16* WqT = p; p += (size_t)QD * QD;        // [N][K]
  __bf16* WkT = p; p += (size_t)QD * CD;
  __bf16* WvT = p; p += (size_t)QD * CD;
  __bf16* WoT = p; p += (size_t)QD * QD;
  __bf16* Qb  = p; p += (size_t)BB * NQ * QD;   // [B*NQ][QD]
  __bf16* Kb  = p; p += (size_t)BB * NK * QD;   // [B*NK][QD]
  __bf16* Vt  = p; p += (size_t)BB * NK * QD;   // [B][H][HD][NK]
  __bf16* Ab  = p; p += (size_t)BB * NQ * QD;   // attention output bf16

  int nx = BB * NQ * QD;  // 16,777,216
  int nc = BB * NK * CD;  //  3,145,728
  cvt_f32_bf16<<<(nx + 255) / 256, 256, 0, stream>>>(x, xb, nx);
  cvt_f32_bf16<<<(nc + 255) / 256, 256, 0, stream>>>(ctx, cb, nc);
  transpose_cvt<<<(QD * QD + 255) / 256, 256, 0, stream>>>(Wq, WqT, QD, QD);
  transpose_cvt<<<(CD * QD + 255) / 256, 256, 0, stream>>>(Wk, WkT, CD, QD);
  transpose_cvt<<<(CD * QD + 255) / 256, 256, 0, stream>>>(Wv, WvT, CD, QD);
  transpose_cvt<<<(QD * QD + 255) / 256, 256, 0, stream>>>(Wo, WoT, QD, QD);

  // Q = x @ Wq : M=B*NQ, N=QD, K=QD  -> (M/16)*(N/64) = 1024*16 tiles
  {
    int tiles = ((BB * NQ) / 16) * (QD / 64);
    gemm_bf16<<<tiles / 8, 256, 0, stream>>>(xb, WqT, Qb, QD / 64, QD, QD, QD, QD, 0);
  }
  // K = ctx @ Wk ; V = ctx @ Wv (transposed store) : M=B*NK, N=QD, K=CD
  {
    int tiles = ((BB * NK) / 16) * (QD / 64);
    gemm_bf16<<<tiles / 8, 256, 0, stream>>>(cb, WkT, Kb, QD / 64, CD, CD, CD, QD, 0);
    gemm_bf16<<<tiles / 8, 256, 0, stream>>>(cb, WvT, Vt, QD / 64, CD, CD, CD, QD, 1);
  }
  // attention
  flash_attn<<<dim3(NQ / 16 / 8, NH, BB), 256, 0, stream>>>(Qb, Kb, Vt, Ab);
  // out = Ab @ Wo + bo (fp32)
  {
    int tiles = ((BB * NQ) / 16) * (QD / 64);
    gemm_out<<<tiles / 8, 256, 0, stream>>>(Ab, WoT, bo, out, QD / 64, QD);
  }
}
